// AdaptiveAttention_39204461478394
// MI455X (gfx1250) — compile-verified, matrix operand order checked by reading, not run on hardware
//
#include <hip/hip_runtime.h>
#include <hip/hip_bf16.h>

// ---------------------------------------------------------------------------
// MHA forward for b=2, n=2048, d=1024, heads=16, head_dim=64 on gfx1250.
// bf16 WMMA (v_wmma_f32_16x16x32_bf16), flash-attention fusion, and
// double-buffered async global->LDS staging of shared operands.
// ---------------------------------------------------------------------------

#define N_SEQ   2048
#define D_MODEL 1024
#define QKV_LD  3072
#define HD      64
#define ROWS    4096          // b * n
#define ATTN_SCALE 0.125f     // 64^-0.5

typedef __attribute__((ext_vector_type(8)))  __bf16 v8bf;
typedef __attribute__((ext_vector_type(16))) __bf16 v16bf;
typedef __attribute__((ext_vector_type(8)))  float  v8f;
typedef int v4i32 __attribute__((vector_size(16)));   // pointee type the async builtin expects

#ifndef __has_builtin
#define __has_builtin(x) 0
#endif

#if defined(__HIP_DEVICE_COMPILE__) && __has_builtin(__builtin_amdgcn_global_load_async_to_lds_b128)
#define HAVE_ASYNC_COPY 1
#endif
#if defined(__HIP_DEVICE_COMPILE__) && __has_builtin(__builtin_amdgcn_s_wait_asynccnt)
#define HAVE_WAIT_ASYNC 1
#endif

// 16-byte global -> LDS copy. Async (ASYNCcnt-tracked) when the toolchain
// exposes the gfx1250 builtin; otherwise a register bounce (still b128+ds).
__device__ __forceinline__ void cp16(const __bf16* g, __bf16* l) {
#if defined(HAVE_ASYNC_COPY)
  __builtin_amdgcn_global_load_async_to_lds_b128(
      (__attribute__((address_space(1))) v4i32*)g,
      (__attribute__((address_space(3))) v4i32*)l, /*offset=*/0, /*cpol=*/0);
#else
  *(v8bf*)l = *(const v8bf*)g;
#endif
}

// Wait until at most 2 of this wave's async copies remain in flight
// (i.e. the previous double-buffer slab has fully landed in LDS).
__device__ __forceinline__ void wait_async_le2() {
#if defined(HAVE_WAIT_ASYNC)
  __builtin_amdgcn_s_wait_asynccnt(2);
#elif defined(HAVE_ASYNC_COPY)
  asm volatile("s_wait_asynccnt 0x2" ::: "memory");
#endif
}

// ---------------------------------------------------------------------------
// WMMA wrapper: D = A(16x32 bf16) * B(32x16 bf16) + C(16x16 f32)
// ---------------------------------------------------------------------------
__device__ __forceinline__ v8f wmma_bf16(v16bf a, v16bf b, v8f c) {
  return __builtin_amdgcn_wmma_f32_16x16x32_bf16(
      /*neg_a=*/false, a, /*neg_b=*/false, b,
      /*c_mod=*/(short)0, c, /*reuse_a=*/false, /*reuse_b=*/false);
}

// ---------------------------------------------------------------------------
// Tile loaders matching the CDNA5 16-bit A/B VGPR layout:
//   lane = {g = lane>>4, m = lane&15}
//   regs 0..3  hold K = g*8 + 0..7        (one contiguous 16-byte run)
//   regs 4..7  hold K = 16 + g*8 + 0..7   (second contiguous 16-byte run)
// Works on global or LDS pointers (generic addressing -> global_/ds_ loads).
// ---------------------------------------------------------------------------
__device__ __forceinline__ v16bf load_tile_rowmajor(const __bf16* __restrict__ base,
                                                    long ld, int row0, int k0, int lane) {
  const int g = (lane >> 4) & 1;
  const int m = lane & 15;
  const __bf16* p = base + (long)(row0 + m) * ld + k0 + g * 8;
  v8bf r0 = *(const v8bf*)(p);
  v8bf r1 = *(const v8bf*)(p + 16);
  v16bf a;
#pragma unroll
  for (int i = 0; i < 8; ++i) { a[i] = r0[i]; a[i + 8] = r1[i]; }
  return a;
}

// B tile from a [K][N] row-major operand (e.g. V): lane owns column n,
// gathers strided K values (row stride `ld`).
__device__ __forceinline__ v16bf load_tile_colmajor(const __bf16* __restrict__ base,
                                                    long ld, int k0, int n0, int lane) {
  const int g = (lane >> 4) & 1;
  const int n = lane & 15;
  const __bf16* p = base + (long)(k0 + g * 8) * ld + n0 + n;
  v16bf b;
#pragma unroll
  for (int i = 0; i < 8; ++i) {
    b[i]     = p[(long)i * ld];
    b[i + 8] = p[(long)(16 + i) * ld];
  }
  return b;
}

// ---------------------------------------------------------------------------
// Conversion kernels (fp32 -> bf16, optionally transposed)
// ---------------------------------------------------------------------------
__global__ void cvt_bf16(const float* __restrict__ in, __bf16* __restrict__ out, long n) {
  long i = (long)blockIdx.x * blockDim.x + threadIdx.x;
  if (i < n) out[i] = (__bf16)in[i];
}

__global__ void cvt_bf16_T(const float* __restrict__ in, __bf16* __restrict__ out,
                           int R, int C) {  // in[R][C] -> out[C][R]
  long i = (long)blockIdx.x * blockDim.x + threadIdx.x;
  if (i < (long)R * C) {
    int r = (int)(i / C), c = (int)(i % C);
    out[(long)c * R + r] = (__bf16)in[i];
  }
}

// ---------------------------------------------------------------------------
// GEMM: Y[M][Nc] = A[M][K] @ WT[Nc][K]^T, bf16 in, f32 accumulate.
// Block = 128 threads = 4 waves, 64x64 output tile. The B slab (64 cols x 32 k,
// 4 KB) is shared by all 4 waves: staged in LDS with double-buffered async
// copies; A tiles stream from global (distinct rows per wave, L2-served).
// ---------------------------------------------------------------------------
__global__ __launch_bounds__(128) void gemm_bf16(const __bf16* __restrict__ A,
                                                 const __bf16* __restrict__ WT,
                                                 __bf16* __restrict__ Y,
                                                 int Nc, int K) {
  __shared__ __align__(16) __bf16 Bs[2][64][32];
  const int tid = threadIdx.x;
  const int lane = tid & 31, wave = tid >> 5;
  const int g = lane >> 4, nl = lane & 15;
  const int row0 = blockIdx.y * 64 + wave * 16;
  const int col0 = blockIdx.x * 64;

  auto stage_B = [&](int buf, int k) {
#pragma unroll
    for (int j = 0; j < 2; ++j) {               // 256 16B chunks / 128 threads
      const int u = tid + j * 128;
      const int r = u >> 2, c = (u & 3) * 8;
      cp16(WT + (long)(col0 + r) * K + k + c, &Bs[buf][r][c]);
    }
  };

  stage_B(0, 0);
  v8f acc[4] = {};
  for (int k = 0; k < K; k += 32) {
    const int cur = (k >> 5) & 1;
    __syncthreads();                            // nxt buffer free to overwrite
    stage_B(cur ^ 1, (k + 32 < K) ? (k + 32) : 0);
    wait_async_le2();                           // cur slab landed
    __syncthreads();                            // ...for every wave
    // prefetch next A tile while this one is consumed
    __builtin_prefetch(A + (long)(row0 + (lane & 15)) * K + ((k + 32 < K) ? k + 32 : 0), 0, 3);
    v16bf a = load_tile_rowmajor(A, K, row0, k, lane);
#pragma unroll
    for (int t = 0; t < 4; ++t) {
      v16bf b = load_tile_rowmajor(&Bs[cur][0][0], 32, t * 16, 0, lane);
      acc[t] = wmma_bf16(a, b, acc[t]);
    }
  }
#pragma unroll
  for (int t = 0; t < 4; ++t)
#pragma unroll
    for (int v = 0; v < 8; ++v)
      Y[(long)(row0 + v + 8 * g) * Nc + col0 + t * 16 + nl] = (__bf16)acc[t][v];
}

// Final projection: f32 output + bias (same staging scheme).
__global__ __launch_bounds__(128) void gemm_bf16_bias_f32(const __bf16* __restrict__ A,
                                                          const __bf16* __restrict__ WT,
                                                          const float* __restrict__ bias,
                                                          float* __restrict__ Y,
                                                          int Nc, int K) {
  __shared__ __align__(16) __bf16 Bs[2][64][32];
  const int tid = threadIdx.x;
  const int lane = tid & 31, wave = tid >> 5;
  const int g = lane >> 4, nl = lane & 15;
  const int row0 = blockIdx.y * 64 + wave * 16;
  const int col0 = blockIdx.x * 64;

  auto stage_B = [&](int buf, int k) {
#pragma unroll
    for (int j = 0; j < 2; ++j) {
      const int u = tid + j * 128;
      const int r = u >> 2, c = (u & 3) * 8;
      cp16(WT + (long)(col0 + r) * K + k + c, &Bs[buf][r][c]);
    }
  };

  stage_B(0, 0);
  v8f acc[4] = {};
  for (int k = 0; k < K; k += 32) {
    const int cur = (k >> 5) & 1;
    __syncthreads();
    stage_B(cur ^ 1, (k + 32 < K) ? (k + 32) : 0);
    wait_async_le2();
    __syncthreads();
    __builtin_prefetch(A + (long)(row0 + (lane & 15)) * K + ((k + 32 < K) ? k + 32 : 0), 0, 3);
    v16bf a = load_tile_rowmajor(A, K, row0, k, lane);
#pragma unroll
    for (int t = 0; t < 4; ++t) {
      v16bf b = load_tile_rowmajor(&Bs[cur][0][0], 32, t * 16, 0, lane);
      acc[t] = wmma_bf16(a, b, acc[t]);
    }
  }
#pragma unroll
  for (int t = 0; t < 4; ++t)
#pragma unroll
    for (int v = 0; v < 8; ++v) {
      const int c = col0 + t * 16 + nl;
      Y[(long)(row0 + v + 8 * g) * Nc + c] = acc[t][v] + bias[c];
    }
}

// ---------------------------------------------------------------------------
// Flash attention. qkv[row][0:1024|1024:2048|2048:3072] = Q|K|V, row=b*2048+n.
// Block = 256 threads = 8 waves; wave owns one 16-query tile of head (b,h).
// The K/V block (32 keys x 64 hd, 8 KB) is shared by all 8 waves: staged in
// LDS with double-buffered async copies. Scores never touch HBM.
// C/D layout: element [v] in lane l -> row (v + 8*(l>=16)), col (l&15).
// ---------------------------------------------------------------------------
__global__ __launch_bounds__(256) void attn_kernel(const __bf16* __restrict__ qkv,
                                                   const int* __restrict__ mask,
                                                   __bf16* __restrict__ attnout) {
  __shared__ __align__(16) __bf16 Ks[2][32][64];
  __shared__ __align__(16) __bf16 Vs[2][32][64];
  __shared__ __align__(16) __bf16 Plds[8][16][32];

  const int tid  = threadIdx.x;
  const int lane = tid & 31;
  const int wave = tid >> 5;
  const int g = lane >> 4, nl = lane & 15;
  const int h  = blockIdx.y;
  const int bb = blockIdx.z;
  const int q0 = (blockIdx.x * 8 + wave) * 16;  // query tile start within seq
  const int rowbase = bb * N_SEQ;

  // Cooperative K/V block stage: 32 rows x 128 B per matrix = 1 chunk each.
  auto stage_kv = [&](int buf, int kb) {
    const int r = tid >> 3, c = (tid & 7) * 8;
    const __bf16* src = qkv + (long)(rowbase + kb + r) * QKV_LD + h * HD + c;
    cp16(src + D_MODEL,     &Ks[buf][r][c]);
    cp16(src + 2 * D_MODEL, &Vs[buf][r][c]);
  };

  // Q tile (16 x 64) as two A tiles (K-steps of 32), kept in registers.
  v16bf qa[2];
#pragma unroll
  for (int ks = 0; ks < 2; ++ks)
    qa[ks] = load_tile_rowmajor(qkv, QKV_LD, rowbase + q0, h * HD + ks * 32, lane);

  v8f o[4] = {};
  float run_mx[8], run_sum[8];
#pragma unroll
  for (int v = 0; v < 8; ++v) { run_mx[v] = -1e30f; run_sum[v] = 0.0f; }

  stage_kv(0, 0);
  for (int kb = 0; kb < N_SEQ; kb += 32) {
    const int cur = (kb >> 5) & 1;
    __syncthreads();                             // nxt buffer free
    stage_kv(cur ^ 1, (kb + 32 < N_SEQ) ? (kb + 32) : 0);
    wait_async_le2();                            // cur K/V landed
    __syncthreads();

    // ---- S = Q @ K^T (16x32 score tile = two 16x16 accumulators), K from LDS
    v8f s0 = {}, s1 = {};
#pragma unroll
    for (int ks = 0; ks < 2; ++ks) {
      v16bf kt0 = load_tile_rowmajor(&Ks[cur][0][0], HD, 0,  ks * 32, lane);
      v16bf kt1 = load_tile_rowmajor(&Ks[cur][0][0], HD, 16, ks * 32, lane);
      s0 = wmma_bf16(qa[ks], kt0, s0);
      s1 = wmma_bf16(qa[ks], kt1, s1);
    }

    // ---- scale + mask (mask is [n][n], broadcast over b,h; resident in L2)
    float p0[8], p1[8], mx[8];
#pragma unroll
    for (int v = 0; v < 8; ++v) {
      const int qrow = q0 + v + 8 * g;
      const int* mrow = mask + (long)qrow * N_SEQ + kb + nl;
      float a0 = (mrow[0]  != 0) ? s0[v] * ATTN_SCALE : -1e9f;
      float a1 = (mrow[16] != 0) ? s1[v] * ATTN_SCALE : -1e9f;
      p0[v] = a0; p1[v] = a1;
      mx[v] = fmaxf(a0, a1);
    }
    // row-max across the 16 lanes of each half (xor 1..8 stays in-half)
#pragma unroll
    for (int v = 0; v < 8; ++v)
#pragma unroll
      for (int m = 1; m < 16; m <<= 1)
        mx[v] = fmaxf(mx[v], __shfl_xor(mx[v], m, 32));

    // ---- online softmax update
    float corr[8];
#pragma unroll
    for (int v = 0; v < 8; ++v) {
      float nm = fmaxf(run_mx[v], mx[v]);
      corr[v] = __expf(run_mx[v] - nm);
      run_mx[v] = nm;
      p0[v] = __expf(p0[v] - nm);
      p1[v] = __expf(p1[v] - nm);
      float rs = p0[v] + p1[v];
#pragma unroll
      for (int m = 1; m < 16; m <<= 1) rs += __shfl_xor(rs, m, 32);
      run_sum[v] = run_sum[v] * corr[v] + rs;
    }
#pragma unroll
    for (int t = 0; t < 4; ++t)
#pragma unroll
      for (int v = 0; v < 8; ++v) o[t][v] *= corr[v];

    // ---- stage P (16x32 bf16) through LDS: C-layout (lane=col) -> A-layout
    // (lane=row). Same-wave DS ops are in-order; fences stop reordering.
#pragma unroll
    for (int v = 0; v < 8; ++v) {
      Plds[wave][v + 8 * g][nl]      = (__bf16)p0[v];
      Plds[wave][v + 8 * g][nl + 16] = (__bf16)p1[v];
    }
    asm volatile("" ::: "memory");
    v16bf pa = load_tile_rowmajor(&Plds[wave][0][0], 32, 0, 0, lane);
    asm volatile("" ::: "memory");

    // ---- O += P @ V  (V 32x64 row-major in LDS -> column-major B reads)
#pragma unroll
    for (int t = 0; t < 4; ++t) {
      v16bf vb = load_tile_colmajor(&Vs[cur][0][0], HD, 0, t * 16, lane);
      o[t] = wmma_bf16(pa, vb, o[t]);
    }
  }

  // ---- epilogue: normalize and scatter to [b, n, h*64 + hd] (bf16)
#pragma unroll
  for (int t = 0; t < 4; ++t)
#pragma unroll
    for (int v = 0; v < 8; ++v) {
      const int qrow = q0 + v + 8 * g;
      float val = o[t][v] / run_sum[v];
      attnout[(long)(rowbase + qrow) * D_MODEL + h * HD + t * 16 + nl] = (__bf16)val;
    }
}

// ---------------------------------------------------------------------------
// Launcher
// ---------------------------------------------------------------------------
extern "C" void kernel_launch(void* const* d_in, const int* in_sizes, int n_in,
                              void* d_out, int out_size, void* d_ws, size_t ws_size,
                              hipStream_t stream) {
  (void)in_sizes; (void)n_in; (void)out_size; (void)ws_size;
  const float* x    = (const float*)d_in[0];  // [2,2048,1024]
  const int*   mask = (const int*)d_in[1];    // [1,1,2048,2048]
  const float* Wqkv = (const float*)d_in[2];  // [1024,3072]
  const float* Wout = (const float*)d_in[3];  // [1024,1024]
  const float* bout = (const float*)d_in[4];  // [1024]

  char* ws = (char*)d_ws;
  __bf16* Xb      = (__bf16*)ws; ws += (size_t)ROWS * D_MODEL * 2;     // 8 MiB
  __bf16* WqkvT   = (__bf16*)ws; ws += (size_t)QKV_LD * D_MODEL * 2;   // 6 MiB
  __bf16* WoutT   = (__bf16*)ws; ws += (size_t)D_MODEL * D_MODEL * 2;  // 2 MiB
  __bf16* qkv     = (__bf16*)ws; ws += (size_t)ROWS * QKV_LD * 2;      // 24 MiB
  __bf16* attnout = (__bf16*)ws;                                       // 8 MiB

  const long nx = (long)ROWS * D_MODEL;
  cvt_bf16<<<(int)((nx + 255) / 256), 256, 0, stream>>>(x, Xb, nx);
  cvt_bf16_T<<<(D_MODEL * QKV_LD + 255) / 256, 256, 0, stream>>>(Wqkv, WqkvT, D_MODEL, QKV_LD);
  cvt_bf16_T<<<(D_MODEL * D_MODEL + 255) / 256, 256, 0, stream>>>(Wout, WoutT, D_MODEL, D_MODEL);

  gemm_bf16<<<dim3(QKV_LD / 64, ROWS / 64), 128, 0, stream>>>(Xb, WqkvT, qkv, QKV_LD, D_MODEL);

  attn_kernel<<<dim3(N_SEQ / 128, 16, 2), 256, 0, stream>>>(qkv, mask, attnout);

  gemm_bf16_bias_f32<<<dim3(D_MODEL / 64, ROWS / 64), 128, 0, stream>>>(
      attnout, WoutT, bout, (float*)d_out, D_MODEL, D_MODEL);
}
